// VQVAE_55611236548731
// MI455X (gfx1250) — compile-verified, matrix operand order checked by reading, not run on hardware
//
#include <hip/hip_runtime.h>
#include <hip/hip_bf16.h>

// ---------------------------------------------------------------------------
// VQ-VAE forward for MI455X (gfx1250).
// - All conv / convT / 1x1 / VQ-distance ops -> v_wmma_f32_16x16x32_bf16.
// - Tap-major implicit GEMM: K = input channels (no div/mod in inner loop),
//   weights repacked to [tap][Cout][Cin] bf16.
// - Input windows staged global->LDS with gfx1250 async DMA
//   (global_load_async_to_lds_b32 + s_wait_asynccnt); fragments from LDS.
// - Templated on (mode, KW, stride, pad): LDS geometry is constexpr.
// ---------------------------------------------------------------------------

typedef __bf16 bf16;
typedef __attribute__((ext_vector_type(16))) __bf16 v16bf;
typedef __attribute__((ext_vector_type(8)))  float  v8f;

union Frag16 { v16bf v; unsigned u[8]; };
union BF2U   { bf16 h; unsigned short s; };

__device__ __forceinline__ unsigned short bf_bits(bf16 x) { BF2U t; t.h = x; return t.s; }

typedef __attribute__((address_space(3))) void as3void;
__device__ __forceinline__ unsigned lds_off(void* p) {
  return (unsigned)(unsigned long long)(as3void*)p;   // generic -> LDS byte offset
}

// gfx1250 async DMA: copy 4 bytes global -> LDS (tracked by ASYNCcnt).
__device__ __forceinline__ void async_g2l_b32(unsigned laddr, const void* g) {
  asm volatile("global_load_async_to_lds_b32 %0, %1, off"
               :: "v"(laddr), "v"((unsigned long long)g) : "memory");
}
__device__ __forceinline__ void wait_async0() {
  asm volatile("s_wait_asynccnt 0x0" ::: "memory");
}

// ---------------------------------------------------------------------------
// Fused conv / convT as tap-major implicit GEMM on WMMA tiles.
//   in : bf16 [B][Cin][Tin];  wt : bf16 [KW][Cout][Cin] (tap-major)
//   y = acc*scale[m]+shift[m]; + addf (f32 residual) + addb (bf16 skip)
//   act 0:none 1:relu 2:leaky0.2
// Block 256 thr = 8 waves; block tile M=32 x N=128 (2 C-tiles per wave,
// shared A fragment); wave tile 16x16; K step = 32 channels.
// MODE 0: conv it = n*STRIDE + kk - PAD
// MODE 1: convT(s2,p1): it = (n+1-kk)/2, valid iff numerator even & >= 0
// ---------------------------------------------------------------------------
template <int MODE, int KW, int STRIDE, int PAD>
__global__ __launch_bounds__(256) void conv_wmma(
    const bf16* __restrict__ in, const bf16* __restrict__ wt,
    const float* __restrict__ scale, const float* __restrict__ shift,
    const float* __restrict__ addf, const bf16* __restrict__ addb,
    bf16* __restrict__ outb, float* __restrict__ outf,
    int Cin, int Cout, int Tin, int Tout, int act)
{
  constexpr int LEN   = (MODE == 0) ? ((127 * STRIDE + KW + 2) & ~1) : 68;
  constexpr int PITCH = LEN + 4;                     // bank stagger
  constexpr int NPAIR = LEN >> 1;
  __shared__ bf16 tile[32 * PITCH];                  // <= 16.9 KB of 320 KB/WGP

  const int lane = threadIdx.x & 31;
  const int wv   = threadIdx.x >> 5;
  const int hf   = lane >> 4;                        // half-wave (ISA 7.12.2)
  const int l16  = lane & 15;
  const int m0   = blockIdx.y * 32 + (wv & 1) * 16;  // Cout tile (this wave)
  const int n0b  = blockIdx.x * 128;                 // Tout tile (block)
  const int bb   = blockIdx.z;

  const long inBase = (long)bb * Cin * Tin;
  // two N subtiles per wave, 64 apart, sharing the A fragment
  int nn[2];
  nn[0] = n0b + (wv >> 1) * 16 + l16;
  nn[1] = nn[0] + 64;

  const int itBase   = (MODE == 0) ? (n0b * STRIDE - PAD) : (n0b / 2 - 1);
  const int itBaseAl = itBase & ~1;                  // 4B-align window

  v8f acc0 = {0.f, 0.f, 0.f, 0.f, 0.f, 0.f, 0.f, 0.f};
  v8f acc1 = {0.f, 0.f, 0.f, 0.f, 0.f, 0.f, 0.f, 0.f};

  // per-lane, per-tap B offsets (LDS elements) and validity, hoisted
  int  boff[2][KW];
  bool bval[2][KW];
#pragma unroll
  for (int j = 0; j < 2; ++j)
#pragma unroll
    for (int kk = 0; kk < KW; ++kk) {
      if (MODE == 0) {
        boff[j][kk] = nn[j] * STRIDE + kk - PAD - itBaseAl;
        bval[j][kk] = true;                          // window zero-padded
      } else {
        const int num = nn[j] + 1 - kk;
        bval[j][kk] = (num >= 0) && ((num & 1) == 0);
        boff[j][kk] = (num >> 1) - itBaseAl;         // OOB its zero in LDS
      }
    }

  const long mrow = (long)(m0 + l16);                // A row (M = m0+l16)

  for (int ci0 = 0; ci0 < Cin; ci0 += 32) {
    // ---- stage input window for channels [ci0, ci0+32) into LDS ----
    for (int p = threadIdx.x; p < 32 * NPAIR; p += 256) {
      const int c  = p / NPAIR;                      // constexpr divisor
      const int j  = p - c * NPAIR;
      const int it = itBaseAl + 2 * j;
      bf16* dst = &tile[c * PITCH + 2 * j];
      const bf16* g = in + inBase + (long)(ci0 + c) * Tin + it;
      if (it >= 0 && it + 1 < Tin) {
        async_g2l_b32(lds_off(dst), g);              // aligned b32 DMA
      } else {                                       // T edge: scalar + zero pad
        dst[0] = (it     >= 0 && it     < Tin) ? g[0] : (bf16)0.f;
        dst[1] = (it + 1 >= 0 && it + 1 < Tin) ? g[1] : (bf16)0.f;
      }
    }
    wait_async0();
    __syncthreads();

    // ---- KW taps x one 32-channel WMMA step each ----
#pragma unroll
    for (int kk = 0; kk < KW; ++kk) {
      const bf16* arow = wt + ((long)kk * Cout + mrow) * Cin + ci0;
      if (ci0 + 32 < Cin) __builtin_prefetch(arow + 32, 0, 3);
      Frag16 af;
#pragma unroll
      for (int v = 0; v < 8; ++v) {
        const int kof = 2 * (v & 3) + ((v >> 2) << 4) + (hf << 3);
        af.u[v] = *(const unsigned*)(arow + kof);    // contiguous channel pair
      }
#pragma unroll
      for (int j = 0; j < 2; ++j) {
        Frag16 bfg;
        const int  off = boff[j][kk];
        const bool ok  = bval[j][kk];
#pragma unroll
        for (int v = 0; v < 8; ++v) {
          const int kof = 2 * (v & 3) + ((v >> 2) << 4) + (hf << 3);
          unsigned short e0 = 0, e1 = 0;
          if (ok) {
            e0 = bf_bits(tile[(kof)     * PITCH + off]);
            e1 = bf_bits(tile[(kof + 1) * PITCH + off]);
          }
          bfg.u[v] = (unsigned)e0 | ((unsigned)e1 << 16);
        }
        if (j == 0)
          acc0 = __builtin_amdgcn_wmma_f32_16x16x32_bf16(false, af.v, false, bfg.v,
                                                         (short)0, acc0, false, false);
        else
          acc1 = __builtin_amdgcn_wmma_f32_16x16x32_bf16(false, af.v, false, bfg.v,
                                                         (short)0, acc1, false, false);
      }
    }
    __syncthreads();   // protect LDS window before next chunk overwrites
  }

#pragma unroll
  for (int j = 0; j < 2; ++j) {
    const v8f& acc = j ? acc1 : acc0;
#pragma unroll
    for (int r = 0; r < 8; ++r) {
      const int m = m0 + r + (hf << 3);              // C/D layout: row r + 8*half
      float y = acc[r];
      if (scale)      y = y * scale[m] + shift[m];
      else if (shift) y += shift[m];
      const long o = (long)bb * Cout * Tout + (long)m * Tout + nn[j];
      if (addf) y += addf[o];
      if (addb) y += (float)addb[o];
      if      (act == 1) y = fmaxf(y, 0.0f);
      else if (act == 2) y = (y > 0.0f) ? y : 0.2f * y;
      if (outf) outf[o] = y;
      if (outb) outb[o] = (bf16)y;
    }
  }
}

// ---------------------------------------------------------------------------
// VQ: per position argmin_k (||c_k||^2 - 2 h·c_k) via WMMA, then gather q.
// h-tile (16 positions x 128 dims) staged once per block via async copies.
// ---------------------------------------------------------------------------
__global__ __launch_bounds__(256) void vq_wmma(
    const bf16* __restrict__ h, const bf16* __restrict__ cb,
    const float* __restrict__ cbf, const float* __restrict__ cnorm,
    bf16* __restrict__ q, float* __restrict__ idx_out,
    int D, int Kcodes, int Tq)
{
  __shared__ bf16 ht[128 * 16];          // [d][p], 4 KB
  __shared__ float sv[256];
  __shared__ int   si[256];
  __shared__ int   sfinal[16];

  const int lane  = threadIdx.x & 31;
  const int wv    = threadIdx.x >> 5;
  const int hf    = lane >> 4;
  const int l16   = lane & 15;
  const int nBase = blockIdx.x * 16;     // global position base (16 | Tq)
  const int bb    = nBase / Tq;
  const int t0    = nBase - bb * Tq;
  const long hB   = (long)bb * D * Tq;

  // stage h[b, :, t0:t0+16] -> LDS (all pairs aligned: t0 multiple of 16)
  for (int e = threadIdx.x; e < D * 8; e += 256) {
    const int d = e >> 3, pj = (e & 7) * 2;
    async_g2l_b32(lds_off(&ht[d * 16 + pj]), h + hB + (long)d * Tq + t0 + pj);
  }
  wait_async0();
  __syncthreads();

  float best = 3.4e38f;
  int   bidx = 0;
  for (int ct = wv; ct < (Kcodes >> 4); ct += 8) {
    v8f acc = {0.f, 0.f, 0.f, 0.f, 0.f, 0.f, 0.f, 0.f};
    const bf16* arow = cb + (long)(ct * 16 + l16) * D;
    for (int kb = 0; kb < D; kb += 32) {
      Frag16 af, bfg;
#pragma unroll
      for (int v = 0; v < 8; ++v) {
        const int kof = 2 * (v & 3) + ((v >> 2) << 4) + (hf << 3);
        af.u[v] = *(const unsigned*)(arow + kb + kof);
        unsigned short e0 = bf_bits(ht[(kb + kof)     * 16 + l16]);
        unsigned short e1 = bf_bits(ht[(kb + kof + 1) * 16 + l16]);
        bfg.u[v] = (unsigned)e0 | ((unsigned)e1 << 16);
      }
      acc = __builtin_amdgcn_wmma_f32_16x16x32_bf16(false, af.v, false, bfg.v,
                                                    (short)0, acc, false, false);
    }
#pragma unroll
    for (int r = 0; r < 8; ++r) {
      const int code = ct * 16 + r + (hf << 3);
      const float s = cnorm[code] - 2.0f * acc[r];
      if (s < best) { best = s; bidx = code; }
    }
  }

  sv[threadIdx.x] = best;
  si[threadIdx.x] = bidx;
  __syncthreads();
  if (threadIdx.x < 16) {
    float bv = 3.4e38f; int bi = 0;
    for (int j = threadIdx.x; j < 256; j += 16)
      if (sv[j] < bv) { bv = sv[j]; bi = si[j]; }
    sfinal[threadIdx.x] = bi;
    idx_out[nBase + threadIdx.x] = (float)bi;
  }
  __syncthreads();
  for (int e = threadIdx.x; e < 16 * D; e += 256) {
    const int p = e / D, d = e - p * D;
    const int code = sfinal[p];
    q[hB + (long)d * Tq + (t0 + p)] = (bf16)cbf[(long)code * D + d];
  }
}

// ------------------------------- prep kernels ------------------------------
__global__ void k_cvt_bf16(const float* __restrict__ s, bf16* __restrict__ d, long n) {
  long i = (long)blockIdx.x * 256 + threadIdx.x;
  if (i < n) d[i] = (bf16)s[i];
}
// repack weights to tap-major bf16 [kw][cout][cin].
// srcT=0: src f32 (cout,cin,kw)  srcT=1: src f32 (cin,cout,kw)
__global__ void k_repack_tap(const float* __restrict__ s, bf16* __restrict__ d,
                             int cin, int cout, int kw, int srcT) {
  long n = (long)cin * cout * kw;
  long i = (long)blockIdx.x * 256 + threadIdx.x;
  if (i < n) {
    int kk  = (int)(i / ((long)cout * cin));
    int rem = (int)(i - (long)kk * cout * cin);
    int co  = rem / cin, ci = rem - co * cin;
    long si = srcT ? (((long)ci * cout + co) * kw + kk)
                   : (((long)co * cin + ci) * kw + kk);
    d[i] = (bf16)s[si];
  }
}
// fold conv bias + BN(eval) into scale/shift
__global__ void k_bnprep(const float* g, const float* b, const float* rm, const float* rv,
                         const float* bias, float* scale, float* shift, int c) {
  int i = blockIdx.x * 256 + threadIdx.x;
  if (i < c) {
    float sc = g[i] * rsqrtf(rv[i] + 1e-5f);
    scale[i] = sc;
    float bi = bias ? bias[i] : 0.0f;
    shift[i] = (bi - rm[i]) * sc + b[i];
  }
}
__global__ void k_cbprep(const float* __restrict__ cb, bf16* __restrict__ cbb,
                         float* __restrict__ cn, int K, int D) {
  int k = blockIdx.x * 256 + threadIdx.x;
  if (k < K) {
    float s = 0.f;
    for (int d = 0; d < D; ++d) {
      float v = cb[(long)k * D + d];
      s += v * v;
      cbb[(long)k * D + d] = (bf16)v;
    }
    cn[k] = s;
  }
}

// ---------------------------------------------------------------------------
// Host orchestration. d_in flattening assumed JAX tree order:
//   0:x 1:codebook, then params with dict keys sorted at every level
//   ('dec' before 'enc'; within cnr: bn{b,g,rm,rv}, conv{b,w}, [res{b,w}];
//    within stack: bn, conv, layers[0..1]).
// ---------------------------------------------------------------------------
extern "C" void kernel_launch(void* const* d_in, const int* in_sizes, int n_in,
                              void* d_out, int out_size, void* d_ws, size_t ws_size,
                              hipStream_t stream)
{
  if (n_in < 154) return;
  const int B = 32, Tq = 512, Dm = 128, KC = 2048;
  auto F = [&](int i) { return (const float*)d_in[i]; };

  char*  base = (char*)d_ws;
  size_t off  = 0;
  auto alloc = [&](size_t bytes) -> char* {
    char* p = base + off;
    off += (bytes + 255) & ~(size_t)255;
    return p;
  };

  const long ACT = 32L * 512 * 512;  // 8,388,608 elems at every stage
  bf16* A0 = (bf16*)alloc(ACT * 2);
  bf16* A1 = (bf16*)alloc(ACT * 2);
  bf16* A2 = (bf16*)alloc(ACT * 2);
  bf16* A3 = (bf16*)alloc(ACT * 2);
  float* RES = (float*)alloc(ACT * 4);           // f32 residual-branch scratch
  bf16*  CBB = (bf16*)alloc((long)KC * Dm * 2);  // codebook bf16
  float* CN  = (float*)alloc((long)KC * 4);      // ||c||^2

  // repack any weight to tap-major bf16 [kw][cout][cin]
  auto rp = [&](int idx, int cin, int cout, int kw, int srcT) -> bf16* {
    long n = (long)cin * cout * kw;
    bf16* d = (bf16*)alloc(n * 2);
    k_repack_tap<<<dim3((unsigned)((n + 255) / 256)), dim3(256), 0, stream>>>(
        F(idx), d, cin, cout, kw, srcT);
    return d;
  };
  struct BN { float* sc; float* sh; };
  auto bnp = [&](int gi, int biasIdx, int c) -> BN {    // gi -> bn.b, gi+1 -> bn.g, ...
    float* sc = (float*)alloc((size_t)c * 4);
    float* sh = (float*)alloc((size_t)c * 4);
    k_bnprep<<<dim3((c + 255) / 256), dim3(256), 0, stream>>>(
        F(gi + 1), F(gi + 0), F(gi + 2), F(gi + 3),
        biasIdx >= 0 ? F(biasIdx) : nullptr, sc, sh, c);
    return {sc, sh};
  };
  // dispatch on the four conv shapes used by the model
  auto conv = [&](const bf16* in, const bf16* w, const float* sc, const float* sh,
                  const float* addf, const bf16* addb, bf16* ob, float* of,
                  int Cin, int Cout, int Tin, int Tout, int KW, int mode, int act) {
    dim3 g(Tout / 128, Cout / 32, B);
    if (mode == 1)
      conv_wmma<1, 4, 2, 1><<<g, dim3(256), 0, stream>>>(in, w, sc, sh, addf, addb, ob, of,
                                                         Cin, Cout, Tin, Tout, act);
    else if (KW == 1)
      conv_wmma<0, 1, 1, 0><<<g, dim3(256), 0, stream>>>(in, w, sc, sh, addf, addb, ob, of,
                                                         Cin, Cout, Tin, Tout, act);
    else if (KW == 3)
      conv_wmma<0, 3, 1, 1><<<g, dim3(256), 0, stream>>>(in, w, sc, sh, addf, addb, ob, of,
                                                         Cin, Cout, Tin, Tout, act);
    else
      conv_wmma<0, 4, 2, 1><<<g, dim3(256), 0, stream>>>(in, w, sc, sh, addf, addb, ob, of,
                                                         Cin, Cout, Tin, Tout, act);
  };

  // ---- weights -> tap-major bf16 (enc) ----
  bf16* Wproj  = rp(153, 64, 128, 3, 0);                  // enc.project.conv.w
  bf16* We1l0  = rp(103, 128, 128, 3, 0);                 // enc.enc1.layers[0].conv.w
  bf16* We1l1  = rp(109, 128, 128, 3, 0);
  bf16* We1f   = rp(97,  128, 128, 3, 0);                 // enc.enc1.conv.w
  bf16* Wd1    = rp(81,  128, 256, 4, 0);                 // enc.down1.conv.w
  bf16* Wd1r   = rp(83,  128, 256, 4, 0);                 // enc.down1.res.w
  bf16* We2l0  = rp(121, 256, 256, 3, 0);
  bf16* We2l1  = rp(127, 256, 256, 3, 0);
  bf16* We2f   = rp(115, 256, 256, 3, 0);
  bf16* Wd2    = rp(89,  256, 512, 4, 0);
  bf16* Wd2r   = rp(91,  256, 512, 4, 0);
  bf16* We3l0  = rp(139, 512, 512, 3, 0);
  bf16* We3l1  = rp(145, 512, 512, 3, 0);
  bf16* We3f   = rp(133, 512, 512, 3, 0);
  bf16* Wpvq   = rp(147, 512, 128, 1, 0);                 // enc.pre_vq.w (1x1)
  // ---- weights -> tap-major bf16 (dec) ----
  bf16* Wavq   = rp(3,   128, 512, 1, 0);                 // dec.aft_vq.w (1x1)
  bf16* Wq1l0  = rp(15,  512, 512, 3, 0);
  bf16* Wq1l1  = rp(21,  512, 512, 3, 0);
  bf16* Wq1f   = rp(9,   512, 512, 3, 0);
  bf16* Wu2    = rp(65,  512, 256, 4, 1);                 // dec.up2.conv.w (cin,cout,k)
  bf16* Wu2r   = rp(67,  512, 256, 4, 1);
  bf16* Wq2l0  = rp(33,  256, 256, 3, 0);
  bf16* Wq2l1  = rp(39,  256, 256, 3, 0);
  bf16* Wq2f   = rp(27,  256, 256, 3, 0);
  bf16* Wu3    = rp(73,  256, 128, 4, 1);
  bf16* Wu3r   = rp(75,  256, 128, 4, 1);
  bf16* Wq3l0  = rp(51,  128, 128, 3, 0);
  bf16* Wq3l1  = rp(57,  128, 128, 3, 0);
  bf16* Wq3f   = rp(45,  128, 128, 3, 0);
  bf16* Wprd   = rp(59,  128, 64, 1, 0);                  // dec.project.w (1x1)

  // ---- BN (+ conv bias) folds ----
  BN bProj = bnp(148, 152, 128);
  BN bE1a = bnp(98, 102, 128), bE1b = bnp(104, 108, 128), bE1f = bnp(92, 96, 128);
  BN bD1  = bnp(76, 80, 256);
  BN bE2a = bnp(116, 120, 256), bE2b = bnp(122, 126, 256), bE2f = bnp(110, 114, 256);
  BN bD2  = bnp(84, 88, 512);
  BN bE3a = bnp(134, 138, 512), bE3b = bnp(140, 144, 512), bE3f = bnp(128, 132, 512);
  BN bQ1a = bnp(10, 14, 512),  bQ1b = bnp(16, 20, 512),  bQ1f = bnp(4, 8, 512);
  BN bU2  = bnp(60, 64, 256);
  BN bQ2a = bnp(28, 32, 256),  bQ2b = bnp(34, 38, 256),  bQ2f = bnp(22, 26, 256);
  BN bU3  = bnp(68, 72, 128);
  BN bQ3a = bnp(46, 50, 128),  bQ3b = bnp(52, 56, 128),  bQ3f = bnp(40, 44, 128);

  k_cbprep<<<dim3((KC + 255) / 256), dim3(256), 0, stream>>>(F(1), CBB, CN, KC, Dm);
  { long n = 32L * 64 * 2048;
    k_cvt_bf16<<<dim3((unsigned)((n + 255) / 256)), dim3(256), 0, stream>>>(F(0), A0, n); }

  // ================================ encoder ================================
  conv(A0, Wproj, bProj.sc, bProj.sh, 0, 0, A1, 0, 64, 128, 2048, 2048, 3, 0, 2);
  // enc1 stack
  conv(A1, We1l0, bE1a.sc, bE1a.sh, 0, 0,  A2, 0, 128, 128, 2048, 2048, 3, 0, 2);
  conv(A2, We1l1, bE1b.sc, bE1b.sh, 0, 0,  A3, 0, 128, 128, 2048, 2048, 3, 0, 2);
  conv(A3, We1f,  bE1f.sc, bE1f.sh, 0, A1, A0, 0, 128, 128, 2048, 2048, 3, 0, 1);
  // down1 (residual branch then fused main)
  conv(A0, Wd1r, 0, F(82), 0, 0, 0, RES, 128, 256, 2048, 1024, 4, 0, 0);
  conv(A0, Wd1,  bD1.sc, bD1.sh, RES, 0, A1, 0, 128, 256, 2048, 1024, 4, 0, 2);
  // enc2 stack
  conv(A1, We2l0, bE2a.sc, bE2a.sh, 0, 0,  A2, 0, 256, 256, 1024, 1024, 3, 0, 2);
  conv(A2, We2l1, bE2b.sc, bE2b.sh, 0, 0,  A3, 0, 256, 256, 1024, 1024, 3, 0, 2);
  conv(A3, We2f,  bE2f.sc, bE2f.sh, 0, A1, A0, 0, 256, 256, 1024, 1024, 3, 0, 1);
  // down2
  conv(A0, Wd2r, 0, F(90), 0, 0, 0, RES, 256, 512, 1024, 512, 4, 0, 0);
  conv(A0, Wd2,  bD2.sc, bD2.sh, RES, 0, A1, 0, 256, 512, 1024, 512, 4, 0, 2);
  // enc3 stack
  conv(A1, We3l0, bE3a.sc, bE3a.sh, 0, 0,  A2, 0, 512, 512, 512, 512, 3, 0, 2);
  conv(A2, We3l1, bE3b.sc, bE3b.sh, 0, 0,  A3, 0, 512, 512, 512, 512, 3, 0, 2);
  conv(A3, We3f,  bE3f.sc, bE3f.sh, 0, A1, A0, 0, 512, 512, 512, 512, 3, 0, 1);
  // pre_vq (plain conv + bias)
  conv(A0, Wpvq, 0, F(146), 0, 0, A1, 0, 512, 128, 512, 512, 1, 0, 0);

  // ================================== VQ ===================================
  float* recon = (float*)d_out;
  float* idxo  = recon + 32L * 64 * 2048;
  vq_wmma<<<dim3((B * Tq) / 16), dim3(256), 0, stream>>>(A1, CBB, F(1), CN, A2, idxo, Dm, KC, Tq);

  // ================================ decoder ================================
  conv(A2, Wavq, 0, F(2), 0, 0, A0, 0, 128, 512, 512, 512, 1, 0, 0);
  // dec1 stack
  conv(A0, Wq1l0, bQ1a.sc, bQ1a.sh, 0, 0,  A1, 0, 512, 512, 512, 512, 3, 0, 2);
  conv(A1, Wq1l1, bQ1b.sc, bQ1b.sh, 0, 0,  A3, 0, 512, 512, 512, 512, 3, 0, 2);
  conv(A3, Wq1f,  bQ1f.sc, bQ1f.sh, 0, A0, A1, 0, 512, 512, 512, 512, 3, 0, 1);
  // up2 (convT residual + fused main)
  conv(A1, Wu2r, 0, F(66), 0, 0, 0, RES, 512, 256, 512, 1024, 4, 1, 0);
  conv(A1, Wu2,  bU2.sc, bU2.sh, RES, 0, A0, 0, 512, 256, 512, 1024, 4, 1, 2);
  // dec2 stack
  conv(A0, Wq2l0, bQ2a.sc, bQ2a.sh, 0, 0,  A1, 0, 256, 256, 1024, 1024, 3, 0, 2);
  conv(A1, Wq2l1, bQ2b.sc, bQ2b.sh, 0, 0,  A3, 0, 256, 256, 1024, 1024, 3, 0, 2);
  conv(A3, Wq2f,  bQ2f.sc, bQ2f.sh, 0, A0, A1, 0, 256, 256, 1024, 1024, 3, 0, 1);
  // up3
  conv(A1, Wu3r, 0, F(74), 0, 0, 0, RES, 256, 128, 1024, 2048, 4, 1, 0);
  conv(A1, Wu3,  bU3.sc, bU3.sh, RES, 0, A0, 0, 256, 128, 1024, 2048, 4, 1, 2);
  // dec3 stack
  conv(A0, Wq3l0, bQ3a.sc, bQ3a.sh, 0, 0,  A1, 0, 128, 128, 2048, 2048, 3, 0, 2);
  conv(A1, Wq3l1, bQ3b.sc, bQ3b.sh, 0, 0,  A3, 0, 128, 128, 2048, 2048, 3, 0, 2);
  conv(A3, Wq3f,  bQ3f.sc, bQ3f.sh, 0, A0, A1, 0, 128, 128, 2048, 2048, 3, 0, 1);
  // final projection -> f32 recon
  conv(A1, Wprd, 0, F(58), 0, 0, 0, recon, 128, 64, 2048, 2048, 1, 0, 0);

  (void)in_sizes; (void)out_size; (void)ws_size;
}
